// SelfAttention_22299470201131
// MI455X (gfx1250) — matrix operand order repaired and checked by reading, not compiled, not run on hardware
//
#include <hip/hip_runtime.h>

typedef _Float16 f16;
typedef __attribute__((ext_vector_type(16))) _Float16 v16h;
typedef __attribute__((ext_vector_type(8)))  _Float16 v8h;
typedef __attribute__((ext_vector_type(4)))  _Float16 v4h;
typedef __attribute__((ext_vector_type(8)))  float    v8f;

constexpr int kEmbed = 512;
constexpr int kHeads = 8;
constexpr int kHD    = 64;   // head dim
constexpr int kN     = 4;    // batch
constexpr int kS     = 2048; // seq len

// ---------------------------------------------------------------------------
// WMMA helpers (CDNA5 gfx1250, wave32): D(16x16,f32) = A(16x32,f16)*B(32x16,f16)+C
// ---------------------------------------------------------------------------
__device__ __forceinline__ v8f wmma_f16(v16h a, v16h b, v8f c) {
  // 8 args: (neg_a, A, neg_b, B, c_mod, C, reuse_a, reuse_b)
  return __builtin_amdgcn_wmma_f32_16x16x32_f16(false, a, false, b, (short)0, c,
                                                false, false);
}

union FragU { v16h v; v8h h[2]; };
union Pack8 { v8h v; f16 e[8]; };

// A-matrix fragment (16x32, row-major source, row stride `stride` halfs).
// ISA layout: lanes 0-15 hold row m=lane, K={hi*8+0..7, 16+hi*8+0..7}, hi=lane>>4.
__device__ __forceinline__ v16h load_a_frag(const f16* p, int stride, int lane) {
  const int m = lane & 15, hi = lane >> 4;
  const f16* row = p + (size_t)m * stride;
  FragU u;
  u.h[0] = *(const v8h*)(row + hi * 8);
  u.h[1] = *(const v8h*)(row + 16 + hi * 8);
  return u.v;
}

// B-matrix fragment (32x16). Source stored as B^T: p[n*stride + k] = B(k,n).
// ISA layout: lane holds column n=lane&15, K={hi*8+0..7, 16+hi*8+0..7}, hi=lane>>4.
__device__ __forceinline__ v16h load_b_frag(const f16* p, int stride, int lane) {
  const int nn = lane & 15, hi = lane >> 4;
  const f16* row = p + (size_t)nn * stride;
  FragU u;
  u.h[0] = *(const v8h*)(row + hi * 8);
  u.h[1] = *(const v8h*)(row + 16 + hi * 8);
  return u.v;
}

// ---------------------------------------------------------------------------
// GEMM: Y = X @ W^T.  X:(8192 x 512), W:(512 x 512) row-major (out,in).
// MODE 0: X f32, Y f16 head-split  [n][h][s][d]
// MODE 1: X f32, Y f16 head-transp [n][h][d][s]   (for V^T)
// MODE 2: X f16, Y f32 row-major   [n*s][e]       (final projection)
// Block: 256 thr (8 waves). Tile 128(M) x 64(N), K-step 32. Wave: 32x32 C.
// ---------------------------------------------------------------------------
template <int MODE>
__global__ __launch_bounds__(256) void gemm_kernel(const void* __restrict__ Ap,
                                                   const float* __restrict__ W,
                                                   void* __restrict__ Yp) {
  __shared__ __align__(16) f16 As[128][40];
  __shared__ __align__(16) f16 Bs[64][40];

  const int tid  = threadIdx.x;
  const int lane = tid & 31;
  const int wave = tid >> 5;
  const int nn   = lane & 15;
  const int hi   = lane >> 4;
  const int rowBase = blockIdx.y * 128;
  const int colBase = blockIdx.x * 64;
  const int wm = (wave >> 1) * 32;  // 0,32,64,96
  const int wn = (wave & 1) * 32;   // 0,32

  v8f c00 = {}, c01 = {}, c10 = {}, c11 = {};

  for (int k0 = 0; k0 < kEmbed; k0 += 32) {
    __syncthreads();
    // ---- stage A tile: 128 x 32 (4096 elems, 16/thread as 4-vectors) ----
    if (MODE == 2) {
      const f16* X = (const f16*)Ap;
      for (int i = 0; i < 4; ++i) {
        int v = tid + i * 256;
        int r = v >> 3, c4 = (v & 7) << 2;
        *(v4h*)&As[r][c4] =
            *(const v4h*)(X + (size_t)(rowBase + r) * kEmbed + k0 + c4);
      }
    } else {
      const float* X = (const float*)Ap;
      for (int i = 0; i < 4; ++i) {
        int v = tid + i * 256;
        int r = v >> 3, c4 = (v & 7) << 2;
        float4 x = *(const float4*)(X + (size_t)(rowBase + r) * kEmbed + k0 + c4);
        v4h hh = {(f16)x.x, (f16)x.y, (f16)x.z, (f16)x.w};
        *(v4h*)&As[r][c4] = hh;
      }
    }
    // ---- stage B tile: rows of W are already B^T layout [n][k] ----
    for (int i = 0; i < 2; ++i) {
      int v = tid + i * 256;
      int r = v >> 3, c4 = (v & 7) << 2;
      float4 x = *(const float4*)(W + (size_t)(colBase + r) * kEmbed + k0 + c4);
      v4h hh = {(f16)x.x, (f16)x.y, (f16)x.z, (f16)x.w};
      *(v4h*)&Bs[r][c4] = hh;
    }
    __syncthreads();

    v16h a0 = load_a_frag(&As[wm][0], 40, lane);
    v16h a1 = load_a_frag(&As[wm + 16][0], 40, lane);
    v16h b0 = load_b_frag(&Bs[wn][0], 40, lane);
    v16h b1 = load_b_frag(&Bs[wn + 16][0], 40, lane);
    c00 = wmma_f16(a0, b0, c00);
    c01 = wmma_f16(a0, b1, c01);
    c10 = wmma_f16(a1, b0, c10);
    c11 = wmma_f16(a1, b1, c11);
  }

  // ---- epilogue: C layout = VGPR r, lane: row = r + 8*hi, col = nn ----
  auto storeTile = [&](v8f c, int mt, int nt) {
    for (int r = 0; r < 8; ++r) {
      int R  = rowBase + wm + mt * 16 + r + 8 * hi;  // 0..8191 (= n*S + s)
      int Cc = colBase + wn + nt * 16 + nn;          // 0..511  (= h*64 + d)
      if (MODE == 2) {
        ((float*)Yp)[(size_t)R * kEmbed + Cc] = c[r];
      } else {
        int n = R >> 11, s = R & (kS - 1);
        int h = Cc >> 6, d = Cc & (kHD - 1);
        size_t idx;
        if (MODE == 0)
          idx = (((size_t)(n * kHeads + h)) * kS + s) * kHD + d;
        else
          idx = (((size_t)(n * kHeads + h)) * kHD + d) * kS + s;
        ((f16*)Yp)[idx] = (f16)c[r];
      }
    }
  };
  storeTile(c00, 0, 0);
  storeTile(c01, 0, 1);
  storeTile(c10, 1, 0);
  storeTile(c11, 1, 1);
}

// ---------------------------------------------------------------------------
// Flash attention (causal), TRANSPOSED formulation:
//   S^T = K . Q^T   (A = K tile [16 keys x 32 d], B = Q rows [q][d])
//   O^T = V^T . P^T (A = V^T tile [16 d x 32 j], B = P^T via LDS bounce)
// C-layout then puts q-columns across lanes and keys inside each lane, so the
// softmax row reduction is in-lane + ONE shfl_xor(16) (vs 4-step butterflies).
// Grid: (S/128, N*H). Block: 256 thr = 8 waves; each wave owns 16 q-rows.
// Qh/Kh: [n][h][s][d] f16.  Vt: [n][h][d][s] f16.  Oh: [n][s][h*64+d] f16.
// ---------------------------------------------------------------------------
__global__ __launch_bounds__(256) void flash_kernel(const f16* __restrict__ Qh,
                                                    const f16* __restrict__ Kh,
                                                    const f16* __restrict__ Vt,
                                                    f16* __restrict__ Oh) {
  __shared__ __align__(16) f16 Pb[8][16][40];  // per-wave P^T scratch as [q][key]

  const int lane = threadIdx.x & 31;
  const int wave = threadIdx.x >> 5;
  const int nn   = lane & 15;
  const int hi   = lane >> 4;
  const int nh   = blockIdx.y;
  const int n    = nh >> 3;
  const int h    = nh & 7;
  const int qw   = blockIdx.x * 128 + wave * 16;  // first q row of this wave
  const int qcol = qw + nn;                       // this lane's q column

  const f16* Qb = Qh + (size_t)nh * kS * kHD;
  const f16* Kb = Kh + (size_t)nh * kS * kHD;
  const f16* Vb = Vt + (size_t)nh * kHD * kS;

  // Preload Q as B-fragments (rows of Q give B^T layout [q][d]); two d-halves.
  const f16* qbase = Qb + (size_t)qw * kHD;
  v16h qb0 = load_b_frag(qbase, kHD, lane);       // d = 0..31
  v16h qb1 = load_b_frag(qbase + 32, kHD, lane);  // d = 32..63

  v8f o0 = {}, o1 = {}, o2 = {}, o3 = {};  // O^T tiles: d = dt*16 + r + 8*hi
  float m_s = -1e30f, l_s = 0.0f;          // per-q-column softmax state
  const float scale = 0.04419417382415922f;  // 1/sqrt(512) per reference

  for (int jb = 0; jb <= qw + 15; jb += 32) {
    if (jb + 32 <= qw + 15)  // prefetch next K tile -> global_prefetch_b8
      __builtin_prefetch(Kb + (size_t)(jb + 32 + nn) * kHD, 0, 1);

    // ---- S^T (32 keys x 16 q): A = K rows, B = Q rows ----
    const f16* kt0 = Kb + (size_t)jb * kHD;
    v16h ka00 = load_a_frag(kt0, kHD, lane);                 // keys 0-15,  d 0-31
    v16h ka01 = load_a_frag(kt0 + 32, kHD, lane);            // keys 0-15,  d 32-63
    v16h ka10 = load_a_frag(kt0 + 16 * kHD, kHD, lane);      // keys 16-31, d 0-31
    v16h ka11 = load_a_frag(kt0 + 16 * kHD + 32, kHD, lane); // keys 16-31, d 32-63

    v8f st0 = {}, st1 = {};
    st0 = wmma_f16(ka00, qb0, st0);
    st0 = wmma_f16(ka01, qb1, st0);
    st1 = wmma_f16(ka10, qb0, st1);
    st1 = wmma_f16(ka11, qb1, st1);

    // ---- scale + causal mask: element (key = jb + kt*16 + r + 8*hi, q = qcol)
    const bool diag = (jb + 31 > qw);  // wave-uniform
    float a0[8], a1[8];
    for (int r = 0; r < 8; ++r) {
      float a = st0[r] * scale;
      float b = st1[r] * scale;
      if (diag) {
        int key = jb + r + 8 * hi;
        if (key > qcol)      a = -1e30f;
        if (key + 16 > qcol) b = -1e30f;
      }
      a0[r] = a;
      a1[r] = b;
    }

    // ---- online softmax: in-lane reduce over 16 keys + 1 cross-half shfl ----
    float mx = fmaxf(a0[0], a1[0]);
    for (int r = 1; r < 8; ++r) mx = fmaxf(mx, fmaxf(a0[r], a1[r]));
    mx = fmaxf(mx, __shfl_xor(mx, 16));  // merge keys of the other half-wave
    float mn  = fmaxf(m_s, mx);
    float fac = __expf(m_s - mn);
    float p0[8], p1[8], rs = 0.0f;
    for (int r = 0; r < 8; ++r) {
      p0[r] = __expf(a0[r] - mn);
      p1[r] = __expf(a1[r] - mn);
      rs += p0[r] + p1[r];
    }
    rs += __shfl_xor(rs, 16);
    l_s = l_s * fac + rs;
    m_s = mn;
    for (int r = 0; r < 8; ++r) {
      o0[r] *= fac; o1[r] *= fac; o2[r] *= fac; o3[r] *= fac;
    }

    // ---- P^T -> LDS as [q][key] (B^T layout), vectorized b128 stores ----
    Pack8 u0, u1;
    for (int r = 0; r < 8; ++r) { u0.e[r] = (f16)p0[r]; u1.e[r] = (f16)p1[r]; }
    *(v8h*)&Pb[wave][nn][8 * hi]      = u0.v;  // keys  8*hi .. 8*hi+7
    *(v8h*)&Pb[wave][nn][16 + 8 * hi] = u1.v;  // keys 16+8*hi .. 16+8*hi+7
    // DS ops are in-order per wave; wait for stores before cross-lane reload.
    asm volatile("s_wait_dscnt 0" ::: "memory");
    v16h pb = load_b_frag(&Pb[wave][0][0], 40, lane);

    // ---- O^T += V^T(16d x 32j) . P^T: V^T rows load directly as A-frags ----
    const f16* vbb = Vb + jb;
    v16h va0 = load_a_frag(vbb + (size_t)0  * kS, kS, lane);
    v16h va1 = load_a_frag(vbb + (size_t)16 * kS, kS, lane);
    v16h va2 = load_a_frag(vbb + (size_t)32 * kS, kS, lane);
    v16h va3 = load_a_frag(vbb + (size_t)48 * kS, kS, lane);
    o0 = wmma_f16(va0, pb, o0);
    o1 = wmma_f16(va1, pb, o1);
    o2 = wmma_f16(va2, pb, o2);
    o3 = wmma_f16(va3, pb, o3);
  }

  // ---- finalize: each lane owns output row s = qw+nn; 8 consecutive d per
  //      tile -> 4 packed v8h stores (global_store_b128). ----
  const float inv = 1.0f / l_s;
  f16* orow = Oh + ((size_t)n * kS + qcol) * kEmbed + h * kHD;
  Pack8 w0, w1, w2, w3;
  for (int r = 0; r < 8; ++r) {
    w0.e[r] = (f16)(o0[r] * inv);
    w1.e[r] = (f16)(o1[r] * inv);
    w2.e[r] = (f16)(o2[r] * inv);
    w3.e[r] = (f16)(o3[r] * inv);
  }
  *(v8h*)(orow + 0 * 16 + 8 * hi) = w0.v;
  *(v8h*)(orow + 1 * 16 + 8 * hi) = w1.v;
  *(v8h*)(orow + 2 * 16 + 8 * hi) = w2.v;
  *(v8h*)(orow + 3 * 16 + 8 * hi) = w3.v;
}

// ---------------------------------------------------------------------------
// Launch. Inputs: queries, keys, values, mask(unused: causal is analytic),
// Wq, Wk, Wv, Wo.  Workspace: Qh | Kh | Vt | Oh, 8 MB f16 each (32 MB total).
// ---------------------------------------------------------------------------
extern "C" void kernel_launch(void* const* d_in, const int* in_sizes, int n_in,
                              void* d_out, int out_size, void* d_ws,
                              size_t ws_size, hipStream_t stream) {
  const float* queries = (const float*)d_in[0];
  const float* keys    = (const float*)d_in[1];
  const float* values  = (const float*)d_in[2];
  const float* Wq      = (const float*)d_in[4];
  const float* Wk      = (const float*)d_in[5];
  const float* Wv      = (const float*)d_in[6];
  const float* Wo      = (const float*)d_in[7];

  const size_t per = (size_t)kN * kHeads * kS * kHD;  // 4M halfs = 8 MB
  f16* Qh = (f16*)d_ws;
  f16* Kh = Qh + per;
  f16* Vt = Kh + per;
  f16* Oh = Vt + per;

  dim3 gg(kEmbed / 64, (kN * kS) / 128);  // (8, 64)
  gemm_kernel<0><<<gg, 256, 0, stream>>>(queries, Wq, Qh);
  gemm_kernel<0><<<gg, 256, 0, stream>>>(keys, Wk, Kh);
  gemm_kernel<1><<<gg, 256, 0, stream>>>(values, Wv, Vt);

  flash_kernel<<<dim3(kS / 128, kN * kHeads), 256, 0, stream>>>(Qh, Kh, Vt, Oh);

  gemm_kernel<2><<<gg, 256, 0, stream>>>(Oh, Wo, d_out);
}
